// classInvariant_58067957842176
// MI455X (gfx1250) — compile-verified, matrix operand order checked by reading, not run on hardware
//
#include <hip/hip_runtime.h>
#include <hip/hip_bf16.h>

typedef _Float16 v8h  __attribute__((ext_vector_type(8)));
typedef _Float16 v16h __attribute__((ext_vector_type(16)));
typedef float    v8f  __attribute__((ext_vector_type(8)));

#define N1 4096
#define N2 4096
#define DD 256          // feature dim
#define DELTA 1.0f
#define EPS 1e-8f

// ---------------------------------------------------------------- init
__global__ void init_acc_kernel(double* acc) { *acc = 0.0; }

// ------------------------------------------------------- row normalize
// One block (256 threads = 8 waves) per row; rows [0,N1) -> mmd1,
// rows [N1, N1+N2) -> mmd2. Writes f16 normalized rows.
__global__ __launch_bounds__(256) void normalize_f16_kernel(
    const float* __restrict__ m1, const float* __restrict__ m2,
    _Float16* __restrict__ h1, _Float16* __restrict__ h2)
{
    const int row = blockIdx.x;
    const float* src;
    _Float16*    dst;
    if (row < N1) { src = m1 + (size_t)row * DD;        dst = h1 + (size_t)row * DD; }
    else          { src = m2 + (size_t)(row - N1) * DD; dst = h2 + (size_t)(row - N1) * DD; }

    const int t = threadIdx.x;
    const float x = src[t];
    float s = x * x;
    #pragma unroll
    for (int off = 16; off > 0; off >>= 1)
        s += __shfl_xor(s, off, 32);

    __shared__ float wsum[8];
    const int wave = t >> 5, lane = t & 31;
    if (lane == 0) wsum[wave] = s;
    __syncthreads();
    float tot = 0.f;
    #pragma unroll
    for (int i = 0; i < 8; ++i) tot += wsum[i];

    const float scale = 1.0f / fmaxf(sqrtf(tot), EPS);
    dst[t] = (_Float16)(x * scale);
}

// ------------------------------------------- WMMA GEMM + hinge + reduce
// Register-blocked: each wave computes a 4x2 grid of 16x16 tiles (64x32
// output) -> 6 panels fetched for 8 tiles of work (6KB/tile vs 16KB
// unblocked). 16-bit A/B VGPR layout (ISA 7.12.2): lane<16 holds
// K{0..7,16..23}, lane>=16 holds K{8..15,24..31} -> two 16B loads each.
#define MT 4   // 16-row tiles per wave in M
#define NT 2   // 16-col tiles per wave in N

__global__ __launch_bounds__(256) void cos_hinge_wmma_kernel(
    const _Float16* __restrict__ h1, const _Float16* __restrict__ h2,
    const int* __restrict__ label1, const int* __restrict__ label2,
    double* __restrict__ acc)
{
    const int wave = threadIdx.x >> 5;
    const int lane = threadIdx.x & 31;
    const int st   = blockIdx.x * 8 + wave;   // 8192 super-tiles (64x32)
    const int sI   = st >> 7;                 // 0..63   (M / 64)
    const int sJ   = st & 127;                // 0..127  (N / 32)
    const int lr   = lane & 15;
    const int hi   = lane >> 4;

    const _Float16* aRow[MT];
    const _Float16* bRow[NT];
    #pragma unroll
    for (int p = 0; p < MT; ++p)
        aRow[p] = h1 + (size_t)(sI * (MT * 16) + p * 16 + lr) * DD + hi * 8;
    #pragma unroll
    for (int q = 0; q < NT; ++q)
        bRow[q] = h2 + (size_t)(sJ * (NT * 16) + q * 16 + lr) * DD + hi * 8;

    v8f c[MT][NT] = {};
    #pragma unroll
    for (int k0 = 0; k0 < DD; k0 += 32) {
        v16h A[MT], B[NT];
        #pragma unroll
        for (int p = 0; p < MT; ++p) {
            v8h lo = *(const v8h*)(aRow[p] + k0);
            v8h hi8 = *(const v8h*)(aRow[p] + k0 + 16);
            #pragma unroll
            for (int i = 0; i < 8; ++i) { A[p][i] = lo[i]; A[p][8 + i] = hi8[i]; }
        }
        #pragma unroll
        for (int q = 0; q < NT; ++q) {
            v8h lo = *(const v8h*)(bRow[q] + k0);
            v8h hi8 = *(const v8h*)(bRow[q] + k0 + 16);
            #pragma unroll
            for (int i = 0; i < 8; ++i) { B[q][i] = lo[i]; B[q][8 + i] = hi8[i]; }
        }
        #pragma unroll
        for (int p = 0; p < MT; ++p)
            #pragma unroll
            for (int q = 0; q < NT; ++q)
                c[p][q] = __builtin_amdgcn_wmma_f32_16x16x32_f16(
                              false, A[p], false, B[q],
                              (short)0, c[p][q], false, false);
    }

    // C layout: vgpr r, lane -> (M = r + 8*hi, N = lane&15)
    int lj[NT];
    #pragma unroll
    for (int q = 0; q < NT; ++q)
        lj[q] = label2[sJ * (NT * 16) + q * 16 + lr];

    float sum = 0.f;
    #pragma unroll
    for (int p = 0; p < MT; ++p) {
        #pragma unroll
        for (int r = 0; r < 8; ++r) {
            const int li = label1[sI * (MT * 16) + p * 16 + r + hi * 8];
            #pragma unroll
            for (int q = 0; q < NT; ++q) {
                const float cs = c[p][q][r];
                sum += (li == lj[q]) ? fmaxf(DELTA - cs, 0.0f) : cs;
            }
        }
    }

    #pragma unroll
    for (int off = 16; off > 0; off >>= 1)
        sum += __shfl_xor(sum, off, 32);

    __shared__ float wsum[8];
    if (lane == 0) wsum[wave] = sum;
    __syncthreads();
    if (threadIdx.x == 0) {
        float tot = 0.f;
        #pragma unroll
        for (int i = 0; i < 8; ++i) tot += wsum[i];
        atomicAdd(acc, (double)tot);   // global_atomic_add_f64
    }
}

// ---------------------------------------------------------------- mean
__global__ void finalize_kernel(const double* acc, float* out) {
    out[0] = (float)(*acc / (double)((long long)N1 * (long long)N2));
}

extern "C" void kernel_launch(void* const* d_in, const int* in_sizes, int n_in,
                              void* d_out, int out_size, void* d_ws, size_t ws_size,
                              hipStream_t stream) {
    const float* mmd1   = (const float*)d_in[0];
    const float* mmd2   = (const float*)d_in[1];
    const int*   label1 = (const int*)d_in[2];
    const int*   label2 = (const int*)d_in[3];
    float*       out    = (float*)d_out;

    char* ws = (char*)d_ws;
    double*   acc = (double*)ws;                                   // 8 B
    _Float16* h1  = (_Float16*)(ws + 1024);                        // 2 MB
    _Float16* h2  = (_Float16*)(ws + 1024 + (size_t)N1 * DD * 2);  // 2 MB

    init_acc_kernel<<<1, 1, 0, stream>>>(acc);
    normalize_f16_kernel<<<N1 + N2, 256, 0, stream>>>(mmd1, mmd2, h1, h2);
    const int supertiles = (N1 / (MT * 16)) * (N2 / (NT * 16));   // 8192
    cos_hinge_wmma_kernel<<<supertiles / 8, 256, 0, stream>>>(h1, h2, label1, label2, acc);
    finalize_kernel<<<1, 1, 0, stream>>>(acc, out);
}
